// HardMiningLoss_45569603010928
// MI455X (gfx1250) — matrix operand order repaired and checked by reading, not compiled, not run on hardware
//
#include <hip/hip_runtime.h>

// ---------------------------------------------------------------------------
// HardMiningLoss for MI455X (gfx1250, wave32, WMMA).
//   sim = X @ X^T with X: [8192, 512] f32 L2-normalized.
//   Compute-bound: 2 passes x 68.7 GFLOP of GEMM via v_wmma_f32_16x16x32_f16.
//   X fits in L2 (16 MB << 192 MB); B column blocks staged in LDS and shared
//   by the 8 waves of a workgroup; A tile held as f16 fragments in VGPRs.
//   All reductions are wave-shuffle / fixed-order tree => deterministic.
// ---------------------------------------------------------------------------

typedef __attribute__((ext_vector_type(16))) _Float16 v16h;
typedef __attribute__((ext_vector_type(8)))  _Float16 v8h;
typedef __attribute__((ext_vector_type(4)))  _Float16 v4h;
typedef __attribute__((ext_vector_type(8)))  float    v8f;

static constexpr int   N      = 8192;
static constexpr int   D      = 512;
static constexpr int   KSTEPS = D / 32;   // 16 WMMA k-steps per 16x16 tile
static constexpr int   CTILES = N / 16;   // 512 column tiles
static constexpr int   BPAD   = 520;      // padded LDS row (halves) vs bank conflicts
static constexpr float MARGIN = 0.1f;

__device__ __forceinline__ v16h cvt16(float4 a, float4 b, float4 c, float4 d) {
  v16h r;
  r[0]=(_Float16)a.x;  r[1]=(_Float16)a.y;  r[2]=(_Float16)a.z;  r[3]=(_Float16)a.w;
  r[4]=(_Float16)b.x;  r[5]=(_Float16)b.y;  r[6]=(_Float16)b.z;  r[7]=(_Float16)b.w;
  r[8]=(_Float16)c.x;  r[9]=(_Float16)c.y;  r[10]=(_Float16)c.z; r[11]=(_Float16)c.w;
  r[12]=(_Float16)d.x; r[13]=(_Float16)d.y; r[14]=(_Float16)d.z; r[15]=(_Float16)d.w;
  return r;
}

// A fragment (16x32 f16, MxK): lane m=lane&15 holds row m; sub=lane>>4 picks
// K sub-chunks: halves 0..7 -> K = ks*32 + 8*sub + h, halves 8..15 -> +16.
__device__ __forceinline__ void load_a_frags(const float* __restrict__ X,
                                             int rowbase, int m, int sub,
                                             v16h* af) {
  const float* rowp = X + (size_t)(rowbase + m) * D;
  #pragma unroll
  for (int ks = 0; ks < KSTEPS; ++ks) {
    const float4* p0 = (const float4*)(rowp + ks * 32 + 8 * sub);
    const float4* p1 = (const float4*)(rowp + ks * 32 + 16 + 8 * sub);
    af[ks] = cvt16(p0[0], p0[1], p1[0], p1[1]);
  }
}

// Cooperative: stage 16 columns x 512 K of X (f32) as f16 rows in LDS.
__device__ __forceinline__ void stage_b(const float* __restrict__ X,
                                        _Float16 (*Bsh)[BPAD], int ct, int tid) {
  const float4* src = (const float4*)(X + (size_t)ct * 16 * D);
  #pragma unroll
  for (int i = 0; i < 8; ++i) {
    int e  = tid + i * 256;     // 16*128 float4 total
    int rr = e >> 7;
    int c4 = e & 127;
    float4 f = src[rr * 128 + c4];
    v4h h;
    h[0] = (_Float16)f.x; h[1] = (_Float16)f.y;
    h[2] = (_Float16)f.z; h[3] = (_Float16)f.w;
    *(v4h*)&Bsh[rr][c4 * 4] = h;
  }
}

// B fragment (32x16 f16, KxN): lane n=lane&15 holds column n; halves 0..7 ->
// K = ks*32 + 8*sub + h, halves 8..15 -> +16 (matches A fragment).
__device__ __forceinline__ v16h load_b_frag(const _Float16 (*Bsh)[BPAD],
                                            int nl, int ks, int sub) {
  const v8h* p0 = (const v8h*)&Bsh[nl][ks * 32 + 8 * sub];
  const v8h* p1 = (const v8h*)&Bsh[nl][ks * 32 + 16 + 8 * sub];
  v8h b0 = p0[0], b1 = p1[0];
  v16h b;
  #pragma unroll
  for (int i = 0; i < 8; ++i) { b[i] = b0[i]; b[8 + i] = b1[i]; }
  return b;
}

__device__ __forceinline__ float wsum16(float v) {
  #pragma unroll
  for (int off = 8; off > 0; off >>= 1) v += __shfl_xor(v, off, 16);
  return v;
}

// ------------------------- pass 1: per-row thresholds ----------------------
__global__ void __launch_bounds__(256)
hm_pass1(const float* __restrict__ X, const int* __restrict__ T,
         float* __restrict__ minpos, float* __restrict__ maxneg) {
  __shared__ _Float16 Bsh[16][BPAD];
  const int tid  = threadIdx.x;
  const int lane = tid & 31;
  const int wave = tid >> 5;
  const int sub  = lane >> 4;
  const int nl   = lane & 15;
  const int rowbase = (blockIdx.x * 8 + wave) * 16;

  v16h af[KSTEPS];
  load_a_frags(X, rowbase, nl, sub, af);

  int tr[8];
  #pragma unroll
  for (int g = 0; g < 8; ++g) tr[g] = T[rowbase + 8 * sub + g];

  float mp[8], mn[8];
  #pragma unroll
  for (int g = 0; g < 8; ++g) { mp[g] = __builtin_inff(); mn[g] = -__builtin_inff(); }

  for (int ct = 0; ct < CTILES; ++ct) {
    __syncthreads();
    stage_b(X, Bsh, ct, tid);
    __syncthreads();
    const int tc = T[ct * 16 + nl];
    v8f acc = {};
    #pragma unroll
    for (int ks = 0; ks < KSTEPS; ++ks) {
      v16h b = load_b_frag(Bsh, nl, ks, sub);
      acc = __builtin_amdgcn_wmma_f32_16x16x32_f16(false, af[ks], false, b,
                                                   (short)0, acc, false, false);
    }
    const int cgl = ct * 16 + nl;
    #pragma unroll
    for (int g = 0; g < 8; ++g) {
      const float s   = acc[g];
      const int   rg  = rowbase + 8 * sub + g;
      const bool same = (tr[g] == tc);
      if (same && (rg != cgl)) mp[g] = fminf(mp[g], s);  // positives, no self
      if (!same)               mn[g] = fmaxf(mn[g], s);  // negatives
    }
  }
  #pragma unroll
  for (int g = 0; g < 8; ++g) {
    float a = mp[g], b = mn[g];
    #pragma unroll
    for (int off = 8; off > 0; off >>= 1) {
      a = fminf(a, __shfl_xor(a, off, 16));
      b = fmaxf(b, __shfl_xor(b, off, 16));
    }
    if (nl == 0) {
      minpos[rowbase + 8 * sub + g] = a;
      maxneg[rowbase + 8 * sub + g] = b;
    }
  }
}

// ------------------------- pass 2: mined sums ------------------------------
__global__ void __launch_bounds__(256)
hm_pass2(const float* __restrict__ X, const int* __restrict__ T,
         const float* __restrict__ minpos, const float* __restrict__ maxneg,
         float* __restrict__ rowloss, float* __restrict__ rowinv,
         float* __restrict__ pall_s, float* __restrict__ pall_c,
         float* __restrict__ nall_s, float* __restrict__ nall_c) {
  __shared__ _Float16 Bsh[16][BPAD];
  const int tid  = threadIdx.x;
  const int lane = tid & 31;
  const int wave = tid >> 5;
  const int sub  = lane >> 4;
  const int nl   = lane & 15;
  const int rowbase = (blockIdx.x * 8 + wave) * 16;

  v16h af[KSTEPS];
  load_a_frags(X, rowbase, nl, sub, af);

  int tr[8];
  float thrN[8], thrP[8];
  #pragma unroll
  for (int g = 0; g < 8; ++g) {
    const int r = rowbase + 8 * sub + g;
    tr[g]   = T[r];
    thrN[g] = minpos[r] - MARGIN;   // keep negatives with sim >  thrN
    thrP[g] = maxneg[r] + MARGIN;   // keep positives with sim <  thrP
  }

  float pc[8] = {}, nc[8] = {}, ps[8] = {}, ns[8] = {};
  float pas[8] = {}, pac[8] = {}, nas[8] = {}, nac[8] = {};

  for (int ct = 0; ct < CTILES; ++ct) {
    __syncthreads();
    stage_b(X, Bsh, ct, tid);
    __syncthreads();
    const int tc = T[ct * 16 + nl];
    v8f acc = {};
    #pragma unroll
    for (int ks = 0; ks < KSTEPS; ++ks) {
      v16h b = load_b_frag(Bsh, nl, ks, sub);
      acc = __builtin_amdgcn_wmma_f32_16x16x32_f16(false, af[ks], false, b,
                                                   (short)0, acc, false, false);
    }
    const int cgl = ct * 16 + nl;
    #pragma unroll
    for (int g = 0; g < 8; ++g) {
      const float s   = acc[g];
      const int   rg  = rowbase + 8 * sub + g;
      const bool same = (tr[g] == tc);
      if (!same) {
        nas[g] += s; nac[g] += 1.0f;                       // full neg stats
        if (s > thrN[g]) { ns[g] += s; nc[g] += 1.0f; }    // mined negatives
      } else if (rg != cgl) {
        pas[g] += s; pac[g] += 1.0f;                       // full pos stats
        if (s < thrP[g]) { ps[g] += 1.0f - s; pc[g] += 1.0f; }  // mined pos
      }
    }
  }
  #pragma unroll
  for (int g = 0; g < 8; ++g) {
    const float pcR  = wsum16(pc[g]),  ncR  = wsum16(nc[g]);
    const float psR  = wsum16(ps[g]),  nsR  = wsum16(ns[g]);
    const float pasR = wsum16(pas[g]), pacR = wsum16(pac[g]);
    const float nasR = wsum16(nas[g]), nacR = wsum16(nac[g]);
    if (nl == 0) {
      const int r = rowbase + 8 * sub + g;
      const float pl   = psR / fmaxf(pcR, 1.0f);
      const float nlv  = nsR / fmaxf(ncR, 1.0f);
      const bool valid = (ncR >= 1.0f);
      rowloss[r] = valid ? (pl + nlv) : 0.0f;
      rowinv[r]  = valid ? 0.0f : 1.0f;
      pall_s[r] = pasR; pall_c[r] = pacR;
      nall_s[r] = nasR; nall_c[r] = nacR;
    }
  }
}

// ------------------------- final deterministic reduce ----------------------
__global__ void __launch_bounds__(256)
hm_finalize(const float* __restrict__ rowloss, const float* __restrict__ rowinv,
            const float* __restrict__ pall_s, const float* __restrict__ pall_c,
            const float* __restrict__ nall_s, const float* __restrict__ nall_c,
            float* __restrict__ out) {
  __shared__ float sl[256], si[256];
  float a = 0.0f, b = 0.0f;
  for (int i = threadIdx.x; i < N; i += 256) { a += rowloss[i]; b += rowinv[i]; }
  sl[threadIdx.x] = a;
  si[threadIdx.x] = b;
  __syncthreads();
  for (int s = 128; s > 0; s >>= 1) {
    if (threadIdx.x < s) {
      sl[threadIdx.x] += sl[threadIdx.x + s];
      si[threadIdx.x] += si[threadIdx.x + s];
    }
    __syncthreads();
  }
  if (threadIdx.x == 0) {
    out[0] = sl[0] / (float)N;                                   // loss
    out[1] = si[0] / (float)N;                                   // prec
    out[2] = pall_s[N - 1] / fmaxf(pall_c[N - 1], 1.0f);          // mean_pos_sim
    out[3] = nall_s[N - 1] / fmaxf(nall_c[N - 1], 1.0f);          // mean_neg_sim
  }
}

// ---------------------------------------------------------------------------
extern "C" void kernel_launch(void* const* d_in, const int* in_sizes, int n_in,
                              void* d_out, int out_size, void* d_ws, size_t ws_size,
                              hipStream_t stream) {
  (void)in_sizes; (void)n_in; (void)out_size; (void)ws_size;
  const float* X = (const float*)d_in[0];   // [8192, 512] f32, L2-normalized
  const int*   T = (const int*)d_in[1];     // [8192] class ids
  float* out = (float*)d_out;               // 4 scalars

  float* ws      = (float*)d_ws;            // 8 * 8192 f32 = 256 KB scratch
  float* minpos  = ws + 0 * N;
  float* maxneg  = ws + 1 * N;
  float* rowloss = ws + 2 * N;
  float* rowinv  = ws + 3 * N;
  float* pall_s  = ws + 4 * N;
  float* pall_c  = ws + 5 * N;
  float* nall_s  = ws + 6 * N;
  float* nall_c  = ws + 7 * N;

  dim3 grid(N / 16 / 8);                    // 64 WGs, 8 waves (row-tiles) each
  dim3 block(256);
  hm_pass1<<<grid, block, 0, stream>>>(X, T, minpos, maxneg);
  hm_pass2<<<grid, block, 0, stream>>>(X, T, minpos, maxneg,
                                       rowloss, rowinv,
                                       pall_s, pall_c, nall_s, nall_c);
  hm_finalize<<<1, 256, 0, stream>>>(rowloss, rowinv,
                                     pall_s, pall_c, nall_s, nall_c, out);
}
